// GhostModule_42932493091466
// MI455X (gfx1250) — matrix operand order repaired and checked by reading, not compiled, not run on hardware
//
#include <hip/hip_runtime.h>

// ---------------- problem constants ----------------
#define B_   32
#define CIN  240
#define CMID 24     // init_c
#define NEWC 216    // init_c * (ratio-1)
#define H_   56
#define W_   56
#define HW   3136   // 56*56
#define KPAD 256    // CIN padded to multiple of 32 for 16x16x32 WMMA
#define PITCH 264   // LDS row pitch in f16 elems (16B-aligned rows, bank-spread)

typedef _Float16 v16h __attribute__((ext_vector_type(16)));
typedef _Float16 h8   __attribute__((ext_vector_type(8)));
typedef float    v8f  __attribute__((ext_vector_type(8)));

// Build a 16-wide f16 fragment from two contiguous 16-byte LDS chunks
// (matches CDNA5 A/B-matrix VGPR layouts, ISA 7.12.2).
__device__ __forceinline__ v16h ldfrag(const _Float16* p0, const _Float16* p1) {
  h8 a = *(const h8*)p0;
  h8 b = *(const h8*)p1;
  return __builtin_shufflevector(a, b, 0,1,2,3,4,5,6,7,8,9,10,11,12,13,14,15);
}

// ============================================================
// Kernel 1: 1x1 conv (GEMM: [24 x 240] * [240 x M]) + BN1 -> y (f32, scratch)
//   Per block: 128 threads = 4 waves; each wave owns a 16-m tile.
//   Weights staged once per block in LDS (f16, padded 32x256).
//   X tile staged per wave as [m][k] f16 so B-fragments are contiguous.
// ============================================================
__global__ __launch_bounds__(128) void k1_conv1x1_bn(
    const float* __restrict__ x, const float* __restrict__ wp,
    const float* __restrict__ g1, const float* __restrict__ b1,
    const float* __restrict__ m1, const float* __restrict__ v1,
    float* __restrict__ y)
{
  __shared__ _Float16 Wlds[32 * PITCH];        // 32 out-ch rows x 256 k (padded)
  __shared__ _Float16 Xlds[4][16 * PITCH];     // per-wave: 16 m rows x 256 k
  __shared__ float s1[CMID], t1[CMID];

  const int t    = threadIdx.x;
  const int wave = t >> 5;
  const int lane = t & 31;

  // ---- stage weights (zero-padded to 32x256) ----
  for (int i = t; i < 32 * 256; i += 128) {
    int r = i >> 8, k = i & 255;
    float v = (r < CMID && k < CIN) ? wp[r * CIN + k] : 0.f;
    Wlds[r * PITCH + k] = (_Float16)v;
  }
  if (t < CMID) {
    float inv = g1[t] * rsqrtf(v1[t] + 1e-5f);
    s1[t] = inv;
    t1[t] = b1[t] - m1[t] * inv;
  }

  // ---- stage this wave's 16-m X tile ----
  // float4 loads: lanes 0..3 cover one 64B k-row (8 k-rows per iteration),
  // 30 global_load_b128 per lane instead of 120 b32 loads.
  const int m0  = (blockIdx.x * 4 + wave) * 16;   // HW%16==0 -> tile stays in one image
  const int b   = m0 / HW;
  const int hw0 = m0 % HW;
  _Float16* xl = &Xlds[wave][0];
  {
    const int j4   = (lane & 3) * 4;   // m offset: 4 consecutive floats
    const int krow = lane >> 2;        // 0..7: k sub-row within the group of 8
    const float* xb = x + (size_t)b * CIN * HW + hw0 + j4;
    for (int i = 0; i < 30; ++i) {
      int k = i * 8 + krow;
      float4 v4 = *(const float4*)(xb + (size_t)k * HW);
      // hide the 12.5KB-strided k walk
      __builtin_prefetch(xb + (size_t)(k + 16) * HW, 0, 1);
      xl[(j4 + 0) * PITCH + k] = (_Float16)v4.x;
      xl[(j4 + 1) * PITCH + k] = (_Float16)v4.y;
      xl[(j4 + 2) * PITCH + k] = (_Float16)v4.z;
      xl[(j4 + 3) * PITCH + k] = (_Float16)v4.w;
    }
    // zero-pad k = 240..255 (16 rows x 16 cols, 8 entries per lane)
    const int j     = lane & 15;
    const int khalf = lane >> 4;
    #pragma unroll
    for (int z = 0; z < 8; ++z)
      xl[j * PITCH + 240 + khalf * 8 + z] = (_Float16)0.f;
  }
  __syncthreads();

  // ---- WMMA main loop: D[24+pad x 16m] = W[.,k] * X[k,.] ----
  v8f acc0 = {};  // out channels 0..15
  v8f acc1 = {};  // out channels 16..23 (+pad)
  const int r  = lane & 15;
  const int hi = lane >> 4;
  #pragma unroll
  for (int kt = 0; kt < KPAD / 32; ++kt) {
    // B fragment (X): lane holds column n=r, two 8-K chunks (kb, kb+16)
    const int kb = kt * 32 + hi * 8;
    v16h bf = ldfrag(&xl[r * PITCH + kb], &xl[r * PITCH + kb + 16]);
    // A fragments (W): lane holds row, two 8-K chunks (ab, ab+16)
    const int ab = kt * 32 + hi * 8;
    v16h a0 = ldfrag(&Wlds[r * PITCH + ab],        &Wlds[r * PITCH + ab + 16]);
    v16h a1 = ldfrag(&Wlds[(16 + r) * PITCH + ab], &Wlds[(16 + r) * PITCH + ab + 16]);
    acc0 = __builtin_amdgcn_wmma_f32_16x16x32_f16(false, a0, false, bf,
                                                  (short)0, acc0, false, false);
    acc1 = __builtin_amdgcn_wmma_f32_16x16x32_f16(false, a1, false, bf,
                                                  (short)0, acc1, false, false);
  }

  // ---- BN1 + store (D: lane -> m column, VGPR v -> channel row) ----
  const int mcol  = lane & 15;
  const int rbase = (lane >> 4) * 8;
  float* yb = y + (size_t)b * CMID * HW + hw0 + mcol;
  #pragma unroll
  for (int v = 0; v < 8; ++v) {
    int ch = rbase + v;                       // 0..15
    yb[(size_t)ch * HW] = acc0[v] * s1[ch] + t1[ch];
  }
  if (lane < 16) {
    #pragma unroll
    for (int v = 0; v < 8; ++v) {
      int ch = 16 + v;                        // 16..23 (lanes>=16 would be pad rows)
      yb[(size_t)ch * HW] = acc1[v] * s1[ch] + t1[ch];
    }
  }
}

// ============================================================
// Kernel 2: depthwise 3x3 + BN2 -> x1 (scratch) and out[:, 0:24]
// ============================================================
__global__ __launch_bounds__(256) void k2_dw_bn(
    const float* __restrict__ y, const float* __restrict__ wdw,
    const float* __restrict__ g2, const float* __restrict__ b2,
    const float* __restrict__ m2, const float* __restrict__ v2,
    float* __restrict__ x1, float* __restrict__ out)
{
  const int hw = blockIdx.x * 256 + threadIdx.x;
  const int c  = blockIdx.y;
  const int b  = blockIdx.z;
  if (hw >= HW) return;
  const int h = hw / W_, w = hw % W_;
  const float* yp = y + ((size_t)b * CMID + c) * HW;

  float acc = 0.f;
  #pragma unroll
  for (int kh = 0; kh < 3; ++kh)
    #pragma unroll
    for (int kw = 0; kw < 3; ++kw) {
      int hh = h + kh - 1, ww = w + kw - 1;
      float xv = (hh >= 0 && hh < H_ && ww >= 0 && ww < W_) ? yp[hh * W_ + ww] : 0.f;
      acc = fmaf(xv, wdw[c * 9 + kh * 3 + kw], acc);
    }
  float inv = g2[c] * rsqrtf(v2[c] + 1e-5f);
  float val = acc * inv + (b2[c] - m2[c] * inv);
  x1 [((size_t)b * CMID + c) * HW + hw] = val;
  out[((size_t)b * 240  + c) * HW + hw] = val;   // concat slice [:, :24]
}

// ============================================================
// Kernel 3: adder (L1-distance dw, ratio 9) + BN3 + ReLU -> out[:, 24:240]
//   Per block c is fixed: stage its 81 taps + 9 BN params in LDS;
//   each thread loads one 3x3 neighborhood and emits 9 channels.
// ============================================================
__global__ __launch_bounds__(256) void k3_adder_bn_relu(
    const float* __restrict__ x1, const float* __restrict__ wadd,
    const float* __restrict__ g3, const float* __restrict__ b3,
    const float* __restrict__ m3, const float* __restrict__ v3,
    float* __restrict__ out)
{
  __shared__ float wl[81];
  __shared__ float s3[9], t3[9];
  const int c = blockIdx.y;
  const int b = blockIdx.z;
  const int t = threadIdx.x;
  if (t < 81) wl[t] = wadd[c * 81 + t];
  if (t < 9) {
    int ch = c * 9 + t;
    float inv = g3[ch] * rsqrtf(v3[ch] + 1e-5f);
    s3[t] = inv;
    t3[t] = b3[ch] - m3[ch] * inv;
  }
  __syncthreads();

  const int hw = blockIdx.x * 256 + t;
  if (hw >= HW) return;
  const int h = hw / W_, w = hw % W_;
  const float* xp = x1 + ((size_t)b * CMID + c) * HW;

  float nb[9];
  #pragma unroll
  for (int kh = 0; kh < 3; ++kh)
    #pragma unroll
    for (int kw = 0; kw < 3; ++kw) {
      int hh = h + kh - 1, ww = w + kw - 1;
      nb[kh * 3 + kw] =
          (hh >= 0 && hh < H_ && ww >= 0 && ww < W_) ? xp[hh * W_ + ww] : 0.f;
    }

  float* op = out + ((size_t)b * 240 + CMID + c * 9) * HW + hw;
  #pragma unroll
  for (int r = 0; r < 9; ++r) {
    float acc = 0.f;
    #pragma unroll
    for (int i = 0; i < 9; ++i) acc += fabsf(nb[i] - wl[r * 9 + i]);
    float val = fmaxf(-acc * s3[r] + t3[r], 0.f);
    op[(size_t)r * HW] = val;
  }
}

// ============================================================
extern "C" void kernel_launch(void* const* d_in, const int* in_sizes, int n_in,
                              void* d_out, int out_size, void* d_ws, size_t ws_size,
                              hipStream_t stream) {
  const float* x    = (const float*)d_in[0];
  const float* wp   = (const float*)d_in[1];
  const float* g1   = (const float*)d_in[2];
  const float* b1   = (const float*)d_in[3];
  const float* m1   = (const float*)d_in[4];
  const float* v1   = (const float*)d_in[5];
  const float* wdw  = (const float*)d_in[6];
  const float* g2   = (const float*)d_in[7];
  const float* b2   = (const float*)d_in[8];
  const float* m2   = (const float*)d_in[9];
  const float* v2   = (const float*)d_in[10];
  const float* wadd = (const float*)d_in[11];
  const float* g3   = (const float*)d_in[12];
  const float* b3   = (const float*)d_in[13];
  const float* m3   = (const float*)d_in[14];
  const float* v3   = (const float*)d_in[15];
  float* out = (float*)d_out;

  // scratch: y (pre-dw, post-BN1) and x1 (post-BN2), each 32*24*3136 f32 (~9.6MB)
  float* y  = (float*)d_ws;
  float* x1 = y + (size_t)B_ * CMID * HW;

  const int mtiles = (B_ * HW) / 16;              // 6272, divisible by 4
  k1_conv1x1_bn<<<mtiles / 4, 128, 0, stream>>>(x, wp, g1, b1, m1, v1, y);

  dim3 gsp((HW + 255) / 256, CMID, B_);           // (13, 24, 32)
  k2_dw_bn<<<gsp, 256, 0, stream>>>(y, wdw, g2, b2, m2, v2, x1, out);
  k3_adder_bn_relu<<<gsp, 256, 0, stream>>>(x1, wadd, g3, b3, m3, v3, out);
}
